// AttentionHead_13709535609163
// MI455X (gfx1250) — matrix operand order repaired and checked by reading, not compiled, not run on hardware
//
#include <hip/hip_runtime.h>

typedef __attribute__((ext_vector_type(16))) __bf16 bf16x16;
typedef __attribute__((ext_vector_type(8)))  __bf16 bf16x8;
typedef __attribute__((ext_vector_type(8)))  float  f32x8;

union ABf { bf16x16 v; bf16x8 h[2]; };

#define B_N   4
#define T_SEQ 4096
#define E_DIM 1024
#define D_H   64

// ---- CDNA5 async global->LDS helpers (ASYNCcnt-tracked DMA into LDS) ----
static __device__ __forceinline__ unsigned lds_off(const void* p) {
    // generic LDS pointer = {shared aperture, 32-bit LDS offset}
    return (unsigned)(unsigned long long)(uintptr_t)p;
}
static __device__ __forceinline__ void async_copy_b128(unsigned loff, const void* g) {
    asm volatile("global_load_async_to_lds_b128 %0, %1, off"
                 :: "v"(loff), "v"(g) : "memory");
}
static __device__ __forceinline__ void wait_async_all() {
    asm volatile("s_wait_asynccnt 0x0" ::: "memory");
}

// ---------------- stage 0: W [E,64] fp32 -> WT [64,E] bf16 ----------------
__global__ void wt_kernel(const float* __restrict__ W, __bf16* __restrict__ WT) {
    int idx = blockIdx.x * blockDim.x + threadIdx.x;   // over E*D
    int e = idx / D_H;
    int d = idx % D_H;
    WT[(size_t)d * E_DIM + e] = (__bf16)W[idx];
}

// ---------------- stage 1: fused QKV projection ----------------
// 8 waves/block = 8 row tiles. W chunks staged in LDS once per block
// (8x L2-traffic cut), double buffered via async-to-LDS DMA. B operands
// prefetched in groups of 6 (distinct live registers) so one LDS-latency
// exposure is amortized over 6 WMMAs.
__global__ __launch_bounds__(256) void qkv_kernel(
    const float* __restrict__ x,
    const __bf16* __restrict__ wtq, const __bf16* __restrict__ wtk,
    const __bf16* __restrict__ wtv,
    const float* __restrict__ bq, const float* __restrict__ bk,
    const float* __restrict__ bv,
    __bf16* __restrict__ qo, __bf16* __restrict__ ko, __bf16* __restrict__ vto)
{
    __shared__ __attribute__((aligned(16))) __bf16 wstage[2][3 * 64 * 32]; // 24KB

    const int tid  = threadIdx.x;
    const int lane = tid & 31;
    const int wave = tid >> 5;
    const int wgid = blockIdx.x * 8 + wave;        // 0..1023 row tiles
    const int b    = wgid >> 8;
    const int t0   = (wgid & 255) << 4;
    const int r16  = lane & 15;
    const int h    = lane >> 4;
    const int kb   = h << 3;

    const __bf16* wts[3] = {wtq, wtk, wtv};

    // cooperative fill of one 3x(64x32) W chunk: 768 x 16B, 3 per thread
    const int wn = tid >> 2;            // 0..63 output column
    const int wp = (tid & 3) * 8;       // element offset within 32-wide K chunk
    auto fill_w = [&](int buf, int e0) {
#pragma unroll
        for (int p = 0; p < 3; ++p)
            async_copy_b128(lds_off(&wstage[buf][p * 2048 + wn * 32 + wp]),
                            wts[p] + (size_t)wn * E_DIM + e0 + wp);
    };

    f32x8 acc[3][4];
#pragma unroll
    for (int p = 0; p < 3; ++p)
#pragma unroll
        for (int nt = 0; nt < 4; ++nt) acc[p][nt] = (f32x8)0.f;

    const float* xrow = x + ((size_t)b * T_SEQ + t0 + r16) * E_DIM;

    fill_w(0, 0);                                   // prologue
    for (int c = 0; c < E_DIM / 32; ++c) {
        const int e0  = c * 32;
        const int buf = c & 1;

        // A operand loads (global, per-wave) issued before the wait to overlap
        f32x8 f0 = *(const f32x8*)(xrow + e0 + kb);
        f32x8 f1 = *(const f32x8*)(xrow + e0 + 16 + kb);

        wait_async_all();                           // my fills for chunk c done
        __syncthreads();                            // everyone's fills visible
        if (c + 1 < E_DIM / 32) fill_w(buf ^ 1, e0 + 32); // overlap with compute

        ABf xa;
#pragma unroll
        for (int i = 0; i < 8; ++i) {
            xa.v[i]     = (__bf16)f0[i];
            xa.v[8 + i] = (__bf16)f1[i];
        }

        // 12 WMMAs; B operands prefetched in groups of 6 live registers
        auto loadW = [&](ABf& w, int i) {
            const int p = i >> 2, nt = i & 3;
            const __bf16* wrow =
                &wstage[buf][p * 2048 + (nt * 16 + r16) * 32 + h * 8];
            w.h[0] = *(const bf16x8*)(wrow);
            w.h[1] = *(const bf16x8*)(wrow + 16);
        };
#pragma unroll
        for (int g = 0; g < 2; ++g) {
            ABf wb[6];
#pragma unroll
            for (int i = 0; i < 6; ++i) loadW(wb[i], g * 6 + i);
#pragma unroll
            for (int i = 0; i < 6; ++i) {
                const int j = g * 6 + i;
                acc[j >> 2][j & 3] = __builtin_amdgcn_wmma_f32_16x16x32_bf16(
                    false, xa.v, false, wb[i].v, (short)0,
                    acc[j >> 2][j & 3], false, false);
            }
        }
        __syncthreads();                            // done consuming buf
    }

    // bias add + store (q,k row-major; v transposed)
#pragma unroll
    for (int nt = 0; nt < 4; ++nt) {
        const int d = nt * 16 + r16;
        const float biq = bq[d], bik = bk[d], biv = bv[d];
#pragma unroll
        for (int r = 0; r < 8; ++r) {
            const int t = t0 + r + h * 8;
            const size_t off = ((size_t)b * T_SEQ + t) * D_H + d;
            qo[off] = (__bf16)(acc[0][nt][r] + biq);
            ko[off] = (__bf16)(acc[1][nt][r] + bik);
            vto[((size_t)b * D_H + d) * T_SEQ + t] = (__bf16)(acc[2][nt][r] + biv);
        }
    }
}

// ---------------- stage 2: flash attention ----------------
// 8 waves/block = 8 query tiles of the same batch; k/vT chunks staged in
// LDS shared by all 8 waves (8x L2 cut), double buffered async DMA.
__global__ __launch_bounds__(256) void attn_kernel(
    const __bf16* __restrict__ qg, const __bf16* __restrict__ kg,
    const __bf16* __restrict__ vtg, float* __restrict__ out)
{
    __shared__ __attribute__((aligned(16))) __bf16 kstage[2][32 * 64]; // 8KB
    __shared__ __attribute__((aligned(16))) __bf16 vstage[2][64 * 32]; // 8KB
    __shared__ __attribute__((aligned(16))) __bf16 pstage[8][16 * 32]; // 8KB

    const int tid  = threadIdx.x;
    const int lane = tid & 31;
    const int wave = tid >> 5;
    const int wgid = blockIdx.x * 8 + wave;        // 0..1023 query tiles
    const int b    = wgid >> 8;                    // all 8 waves share b
    const int t0   = (wgid & 255) << 4;
    const int r16  = lane & 15;
    const int h    = lane >> 4;
    const int kb   = h << 3;
    const float scale = 0.125f;                    // 1/sqrt(64)

    // cooperative k/vT chunk fill: 2 x 16B per thread
    const int sk = tid >> 3, pk = (tid & 7) * 8;   // k: 32 rows x 64
    const int dv = tid >> 2, pv = (tid & 3) * 8;   // vT: 64 rows x 32
    auto fill_kv = [&](int buf, int j0) {
        async_copy_b128(lds_off(&kstage[buf][sk * 64 + pk]),
                        kg + ((size_t)b * T_SEQ + j0 + sk) * D_H + pk);
        async_copy_b128(lds_off(&vstage[buf][dv * 32 + pv]),
                        vtg + ((size_t)b * D_H + dv) * T_SEQ + j0 + pv);
    };

    // Q as A-operand for both d-chunks (held in registers all iterations)
    ABf qa[2];
    const __bf16* qrow = qg + ((size_t)b * T_SEQ + t0 + r16) * D_H;
#pragma unroll
    for (int c = 0; c < 2; ++c) {
        qa[c].h[0] = *(const bf16x8*)(qrow + c * 32 + kb);
        qa[c].h[1] = *(const bf16x8*)(qrow + c * 32 + 16 + kb);
    }

    f32x8 o[4];
#pragma unroll
    for (int nt = 0; nt < 4; ++nt) o[nt] = (f32x8)0.f;
    float m[8], l[8];
#pragma unroll
    for (int r = 0; r < 8; ++r) { m[r] = -1e30f; l[r] = 0.f; }

    fill_kv(0, 0);                                  // prologue
    for (int cj = 0; cj < T_SEQ / 32; ++cj) {
        const int buf = cj & 1;

        wait_async_all();                           // my fills for chunk cj done
        __syncthreads();                            // everyone's fills visible
        if (cj + 1 < T_SEQ / 32) fill_kv(buf ^ 1, (cj + 1) * 32);

        // scores: prefetch all 4 K operands (distinct registers), then 4 WMMAs
        ABf kbf[4];
#pragma unroll
        for (int i = 0; i < 4; ++i) {
            const int c = i >> 1, t = i & 1;
            const __bf16* kr = &kstage[buf][(t * 16 + r16) * 64 + c * 32 + h * 8];
            kbf[i].h[0] = *(const bf16x8*)kr;
            kbf[i].h[1] = *(const bf16x8*)(kr + 16);
        }
        f32x8 s[2] = {(f32x8)0.f, (f32x8)0.f};
#pragma unroll
        for (int i = 0; i < 4; ++i) {
            s[i & 1] = __builtin_amdgcn_wmma_f32_16x16x32_bf16(
                false, qa[i >> 1].v, false, kbf[i].v, (short)0,
                s[i & 1], false, false);
        }

        // hoist vT B-operands: overlaps LDS latency with softmax VALU work
        ABf vb[4];
#pragma unroll
        for (int nt = 0; nt < 4; ++nt) {
            const __bf16* vr = &vstage[buf][(nt * 16 + r16) * 32 + h * 8];
            vb[nt].h[0] = *(const bf16x8*)vr;
            vb[nt].h[1] = *(const bf16x8*)(vr + 16);
        }

        // online softmax; rows live in (VGPR r, lane-half) -> 16-lane butterflies
#pragma unroll
        for (int r = 0; r < 8; ++r) {
            float a0 = s[0][r] * scale, a1 = s[1][r] * scale;
            float t = fmaxf(a0, a1);
#pragma unroll
            for (int mk = 1; mk < 16; mk <<= 1)
                t = fmaxf(t, __shfl_xor(t, mk, 32));
            const float mn    = fmaxf(m[r], t);
            const float alpha = __expf(m[r] - mn);
            m[r] = mn;
            float p0 = __expf(a0 - mn), p1 = __expf(a1 - mn);
            float rs = p0 + p1;
#pragma unroll
            for (int mk = 1; mk < 16; mk <<= 1)
                rs += __shfl_xor(rs, mk, 32);
            l[r] = l[r] * alpha + rs;
#pragma unroll
            for (int nt = 0; nt < 4; ++nt) o[nt][r] *= alpha;
            pstage[wave][(r + h * 8) * 32 + r16]      = (__bf16)p0;
            pstage[wave][(r + h * 8) * 32 + 16 + r16] = (__bf16)p1;
        }
        asm volatile("s_wait_dscnt 0x0" ::: "memory");

        // reload P in A-operand layout (16x32)
        ABf pa;
        pa.h[0] = *(const bf16x8*)&pstage[wave][r16 * 32 + kb];
        pa.h[1] = *(const bf16x8*)&pstage[wave][r16 * 32 + 16 + kb];

        // O += P * V (B-operands already resident in registers)
#pragma unroll
        for (int nt = 0; nt < 4; ++nt) {
            o[nt] = __builtin_amdgcn_wmma_f32_16x16x32_bf16(
                false, pa.v, false, vb[nt].v, (short)0, o[nt], false, false);
        }
        __syncthreads();                            // done consuming buf
    }

    // normalize and store fp32 output
#pragma unroll
    for (int nt = 0; nt < 4; ++nt) {
#pragma unroll
        for (int r = 0; r < 8; ++r) {
            const float inv = 1.f / l[r];
            out[((size_t)b * T_SEQ + t0 + r + h * 8) * D_H + nt * 16 + r16] =
                o[nt][r] * inv;
        }
    }
}

extern "C" void kernel_launch(void* const* d_in, const int* in_sizes, int n_in,
                              void* d_out, int out_size, void* d_ws, size_t ws_size,
                              hipStream_t stream) {
    (void)in_sizes; (void)n_in; (void)out_size; (void)ws_size;
    const float* x  = (const float*)d_in[0];
    const float* Wq = (const float*)d_in[1];
    const float* bq = (const float*)d_in[2];
    const float* Wk = (const float*)d_in[3];
    const float* bk = (const float*)d_in[4];
    const float* Wv = (const float*)d_in[5];
    const float* bv = (const float*)d_in[6];
    float* out = (float*)d_out;

    // workspace layout (bf16): 3x WT [64,E], q [B,T,64], k [B,T,64], vT [B,64,T]
    __bf16* wtq = (__bf16*)d_ws;
    __bf16* wtk = wtq + (size_t)D_H * E_DIM;
    __bf16* wtv = wtk + (size_t)D_H * E_DIM;
    __bf16* q   = wtv + (size_t)D_H * E_DIM;
    __bf16* k   = q   + (size_t)B_N * T_SEQ * D_H;
    __bf16* vT  = k   + (size_t)B_N * T_SEQ * D_H;

    const int wtBlocks = (E_DIM * D_H) / 256;
    wt_kernel<<<wtBlocks, 256, 0, stream>>>(Wq, wtq);
    wt_kernel<<<wtBlocks, 256, 0, stream>>>(Wk, wtk);
    wt_kernel<<<wtBlocks, 256, 0, stream>>>(Wv, wtv);

    // 1024 row/query tiles, 8 waves per block
    qkv_kernel<<<128, 256, 0, stream>>>(x, wtq, wtk, wtv, bq, bk, bv, q, k, vT);
    attn_kernel<<<128, 256, 0, stream>>>(q, k, vT, out);
}
